// AttnConv2d_10539849744482
// MI455X (gfx1250) — matrix operand order repaired and checked by reading, not compiled, not run on hardware
//
#include <hip/hip_runtime.h>
#include <math.h>

// ---------------- types / helpers ----------------

typedef __attribute__((ext_vector_type(16))) __bf16 v16bf;
typedef __attribute__((ext_vector_type(8)))  float  v8f;

union Frag {
    v16bf v;
    unsigned short us[16];
    uint4 q[2];
};

__device__ __forceinline__ unsigned short f2bf(float f) {
    unsigned u = __float_as_uint(f);
    unsigned r = u + 0x7FFFu + ((u >> 16) & 1u);   // round-to-nearest-even
    return (unsigned short)(r >> 16);
}

// Problem constants
#define BB   8
#define CIN  256
#define COUT 256
#define HH   96
#define WW   96
#define HW   (HH * WW)        // 9216
#define KT   2304             // Cin*3*3, GEMM K for convs / final
#define LL   1024             // (96/3)^2 non-overlapping 3x3 blocks
#define NATT (BB * COUT * KT) // 4,718,592 attn elements

#define WMMA_BF16(A, B, C) \
    __builtin_amdgcn_wmma_f32_16x16x32_bf16(false, (A), false, (B), (short)0, (C), false, false)

// ---------------- kernels ----------------

__global__ __launch_bounds__(256) void cvt_f32_bf16(const float* __restrict__ s,
                                                    unsigned short* __restrict__ d, int n) {
    int stride = gridDim.x * blockDim.x;
    for (int i = blockIdx.x * blockDim.x + threadIdx.x; i < n; i += stride)
        d[i] = f2bf(s[i]);
}

// Materialize im2col (k=3, stride=1, pad=1) transposed: P[b][n][k] bf16, k = c*9+ky*3+kx.
// Stored n-major so WMMA B fragments are contiguous 32B loads per lane.
__global__ __launch_bounds__(256) void im2col_bf16(const float* __restrict__ X,       // [B][256][96][96] f32
                                                   unsigned short* __restrict__ P) {  // [B][9216][2304] bf16
    const int b = blockIdx.y;
    const int n = blockIdx.x;
    const int y = n / WW, x = n - y * WW;
    const float* Xb = X + (size_t)b * (CIN * HW);
    unsigned short* row = P + ((size_t)b * HW + n) * KT;
    for (int k = threadIdx.x; k < KT; k += 256) {
        int c  = k / 9;
        int rp = k - c * 9;
        int ky = rp / 3;
        int kx = rp - ky * 3;
        int sy = y + ky - 1, sx = x + kx - 1;
        float v = 0.0f;
        if ((unsigned)sy < (unsigned)HH && (unsigned)sx < (unsigned)WW)
            v = Xb[(c * HH + sy) * WW + sx];
        row[k] = f2bf(v);
    }
}

// Dense bf16 WMMA GEMM: out[o][n] = sum_k A[o][k] * Bt[n][k]
//   A:  row-major [M][K] (optionally per-batch via aBatchStride)
//   Bt: [b][N][K] (transposed im2col) -> B fragment = contiguous 32B per lane
// Block: 256 threads = 8 waves (2 M x 4 N); wave tile 64x32 (4x2 WMMA accs); block tile 128x128.
// mode 0: write bf16 rearranged as [b][p][c][l] (stride-3 unfold re-index of conv output)
// mode 1: write f32 [b][o][HW] (final output)
__global__ __launch_bounds__(256) void gemm_bf16_wmma(
    const unsigned short* __restrict__ A,
    const unsigned short* __restrict__ Bt,
    unsigned short* __restrict__ outRe,
    float* __restrict__ outF,
    int mode, int aBatchStride)
{
    const int b      = blockIdx.z;
    const int lane   = threadIdx.x & 31;
    const int wave   = threadIdx.x >> 5;
    const int laneHi = lane >> 4;
    const int lane15 = lane & 15;
    const int m0 = blockIdx.y * 128 + (wave >> 2) * 64;   // wave M base (4 subtiles)
    const int n0 = blockIdx.x * 128 + (wave & 3) * 32;    // wave N base (2 subtiles)

    const unsigned short* Abase = A + (size_t)b * aBatchStride;
    const unsigned short* Bbase = Bt + (size_t)b * HW * KT;

    const unsigned short* arow[4];
    #pragma unroll
    for (int i = 0; i < 4; ++i)
        arow[i] = Abase + (size_t)(m0 + i * 16 + lane15) * KT;
    const unsigned short* brow[2];
    #pragma unroll
    for (int j = 0; j < 2; ++j)
        brow[j] = Bbase + (size_t)(n0 + j * 16 + lane15) * KT;

    v8f acc[4][2] = {};
    for (int k0 = 0; k0 < KT; k0 += 32) {
        __builtin_prefetch(arow[0] + k0 + 256, 0, 1);   // global_prefetch_b8
        __builtin_prefetch(brow[0] + k0 + 256, 0, 1);
        Frag fa[4], fb[2];
        #pragma unroll
        for (int i = 0; i < 4; ++i) {
            const uint4* pa = reinterpret_cast<const uint4*>(arow[i] + k0 + laneHi * 16);
            fa[i].q[0] = pa[0];
            fa[i].q[1] = pa[1];   // contiguous 32B
        }
        #pragma unroll
        for (int j = 0; j < 2; ++j) {
            const uint4* pb = reinterpret_cast<const uint4*>(brow[j] + k0 + laneHi * 16);
            fb[j].q[0] = pb[0];
            fb[j].q[1] = pb[1];   // contiguous 32B
        }
        #pragma unroll
        for (int i = 0; i < 4; ++i)
            #pragma unroll
            for (int j = 0; j < 2; ++j)
                acc[i][j] = WMMA_BF16(fa[i].v, fb[j].v, acc[i][j]);
    }

    if (mode == 0) {
        #pragma unroll
        for (int j = 0; j < 2; ++j) {
            int n  = n0 + j * 16 + lane15;
            int y  = n / WW, x = n - y * WW;
            int by = y / 3, dy = y - by * 3;
            int bx = x / 3, dx = x - bx * 3;
            int p = dy * 3 + dx, l = by * 32 + bx;
            unsigned short* dst = outRe + ((size_t)(b * 9 + p) * COUT) * LL + l;
            #pragma unroll
            for (int i = 0; i < 4; ++i) {
                #pragma unroll
                for (int r = 0; r < 8; ++r) {
                    int o = m0 + i * 16 + r + (laneHi << 3);
                    dst[(size_t)o * LL] = f2bf(acc[i][j][r]);
                }
            }
        }
    } else {
        #pragma unroll
        for (int j = 0; j < 2; ++j) {
            int n = n0 + j * 16 + lane15;
            #pragma unroll
            for (int i = 0; i < 4; ++i) {
                #pragma unroll
                for (int r = 0; r < 8; ++r) {
                    int o = m0 + i * 16 + r + (laneHi << 3);
                    outF[((size_t)(b * COUT + o)) * HW + n] = acc[i][j][r];
                }
            }
        }
    }
}

// Per-(b,p) attention GEMMs:
//   a1[o][c] = sum_l q1[b][p][o][l] * key[b][p][c][l]  (a2 with q2; key B-fragments shared)
//   attn_raw[b][o][c*9+p] = (a1 + pos1) * (a2 + pos2)
// Block: 8 waves (2 M x 4 N); wave tile 32x32 with dual outputs -> 8 WMMAs/K-step; block 64x128.
__global__ __launch_bounds__(256) void attn_gemm_wmma(
    const unsigned short* __restrict__ keyArr,  // [B][9][256][1024]
    const unsigned short* __restrict__ q1Arr,
    const unsigned short* __restrict__ q2Arr,
    const float* __restrict__ pos1,             // [256][2304]
    const float* __restrict__ pos2,
    float* __restrict__ attn)                   // [B][256][2304]
{
    const int b = blockIdx.z;
    const int p = blockIdx.y;
    const int mt = blockIdx.x >> 1, nt = blockIdx.x & 1;
    const int lane   = threadIdx.x & 31;
    const int wave   = threadIdx.x >> 5;
    const int laneHi = lane >> 4;
    const int lane15 = lane & 15;
    const int o0 = mt * 64 + (wave >> 2) * 32;
    const int c0 = nt * 128 + (wave & 3) * 32;

    const size_t plane = (size_t)(b * 9 + p) * COUT * LL;
    const unsigned short* q1row[2];
    const unsigned short* q2row[2];
    const unsigned short* krow[2];
    #pragma unroll
    for (int i = 0; i < 2; ++i) {
        q1row[i] = q1Arr + plane + (size_t)(o0 + i * 16 + lane15) * LL;
        q2row[i] = q2Arr + plane + (size_t)(o0 + i * 16 + lane15) * LL;
    }
    #pragma unroll
    for (int j = 0; j < 2; ++j)
        krow[j] = keyArr + plane + (size_t)(c0 + j * 16 + lane15) * LL;

    v8f a1[2][2] = {}, a2[2][2] = {};
    for (int k0 = 0; k0 < LL; k0 += 32) {
        __builtin_prefetch(krow[0] + k0 + 256, 0, 1);
        Frag f1[2], f2[2], fk[2];
        #pragma unroll
        for (int i = 0; i < 2; ++i) {
            const uint4* p1 = reinterpret_cast<const uint4*>(q1row[i] + k0 + laneHi * 16);
            f1[i].q[0] = p1[0]; f1[i].q[1] = p1[1];
            const uint4* p2 = reinterpret_cast<const uint4*>(q2row[i] + k0 + laneHi * 16);
            f2[i].q[0] = p2[0]; f2[i].q[1] = p2[1];
        }
        #pragma unroll
        for (int j = 0; j < 2; ++j) {
            const uint4* pk = reinterpret_cast<const uint4*>(krow[j] + k0 + laneHi * 16);
            fk[j].q[0] = pk[0]; fk[j].q[1] = pk[1];
        }
        #pragma unroll
        for (int i = 0; i < 2; ++i)
            #pragma unroll
            for (int j = 0; j < 2; ++j) {
                a1[i][j] = WMMA_BF16(f1[i].v, fk[j].v, a1[i][j]);
                a2[i][j] = WMMA_BF16(f2[i].v, fk[j].v, a2[i][j]);
            }
    }

    #pragma unroll
    for (int i = 0; i < 2; ++i) {
        #pragma unroll
        for (int j = 0; j < 2; ++j) {
            int c = c0 + j * 16 + lane15;
            #pragma unroll
            for (int r = 0; r < 8; ++r) {
                int o = o0 + i * 16 + r + (laneHi << 3);
                int f = c * 9 + p;
                float v1 = a1[i][j][r] + pos1[o * KT + f];
                float v2 = a2[i][j][r] + pos2[o * KT + f];
                attn[((size_t)(b * COUT + o)) * KT + f] = v1 * v2;
            }
        }
    }
}

__global__ void zero2(float* s) {
    if (threadIdx.x < 2) s[threadIdx.x] = 0.0f;
}

__global__ __launch_bounds__(256) void reduce_sums(const float* __restrict__ a, int n,
                                                   float* __restrict__ s) {
    __shared__ float sh0[256];
    __shared__ float sh1[256];
    float ls = 0.0f, lq = 0.0f;
    int stride = gridDim.x * blockDim.x;
    for (int i = blockIdx.x * blockDim.x + threadIdx.x; i < n; i += stride) {
        float v = a[i];
        ls += v;
        lq += v * v;
    }
    sh0[threadIdx.x] = ls;
    sh1[threadIdx.x] = lq;
    __syncthreads();
    for (int st = 128; st > 0; st >>= 1) {
        if ((int)threadIdx.x < st) {
            sh0[threadIdx.x] += sh0[threadIdx.x + st];
            sh1[threadIdx.x] += sh1[threadIdx.x + st];
        }
        __syncthreads();
    }
    if (threadIdx.x == 0) {
        atomicAdd(&s[0], sh0[0]);   // global_atomic_add_f32
        atomicAdd(&s[1], sh1[0]);
    }
}

__global__ __launch_bounds__(256) void normalize_attn(const float* __restrict__ a,
                                                      const float* __restrict__ s,
                                                      const float* __restrict__ km_p,
                                                      unsigned short* __restrict__ out, int n) {
    float N   = (float)n;
    float mu  = s[0] / N;
    float var = (s[1] - N * mu * mu) / (N - 1.0f);   // unbiased (torch ddof=1)
    float sd  = sqrtf(fmaxf(var, 0.0f));
    float km  = km_p[0];
    float inv = 1.0f / (sd + 1e-4f);
    int stride = gridDim.x * blockDim.x;
    for (int i = blockIdx.x * blockDim.x + threadIdx.x; i < n; i += stride) {
        float v = a[i];
        out[i] = f2bf(km * v + (v - mu) * inv);
    }
}

// ---------------- launcher ----------------

extern "C" void kernel_launch(void* const* d_in, const int* in_sizes, int n_in,
                              void* d_out, int out_size, void* d_ws, size_t ws_size,
                              hipStream_t stream) {
    (void)in_sizes; (void)n_in; (void)out_size; (void)ws_size;

    const float* x1    = (const float*)d_in[0];
    const float* x2    = (const float*)d_in[1];
    const float* key_w = (const float*)d_in[2];
    const float* q1_w  = (const float*)d_in[3];
    const float* q2_w  = (const float*)d_in[4];
    const float* pos1  = (const float*)d_in[5];
    const float* pos2  = (const float*)d_in[6];
    const float* km    = (const float*)d_in[7];
    float* out = (float*)d_out;

    const size_t SP   = (size_t)BB * HW * KT;        // im2col elems per input (169,869,312)
    const size_t SW   = (size_t)COUT * KT;           // 589,824
    const size_t SARR = (size_t)BB * 9 * COUT * LL;  // 18,874,368

    unsigned short* P1     = (unsigned short*)d_ws;          // im2col(x1) [B][n][k]
    unsigned short* P2     = P1 + SP;                        // im2col(x2)
    unsigned short* kwbf   = P2 + SP;
    unsigned short* q1wbf  = kwbf + SW;
    unsigned short* q2wbf  = q1wbf + SW;
    unsigned short* keyArr = q2wbf + SW;
    unsigned short* q1Arr  = keyArr + SARR;
    unsigned short* q2Arr  = q1Arr + SARR;
    float*          attnRaw = (float*)(q2Arr + SARR);
    unsigned short* attnBf  = (unsigned short*)(attnRaw + NATT);
    float*          sums    = (float*)(attnBf + NATT);

    // 1) materialize transposed im2col patches (bf16) + convert weights
    dim3 gIm(HW, BB);
    im2col_bf16<<<gIm, 256, 0, stream>>>(x1, P1);
    im2col_bf16<<<gIm, 256, 0, stream>>>(x2, P2);
    cvt_f32_bf16<<<512, 256, 0, stream>>>(key_w, kwbf, (int)SW);
    cvt_f32_bf16<<<512, 256, 0, stream>>>(q1_w, q1wbf, (int)SW);
    cvt_f32_bf16<<<512, 256, 0, stream>>>(q2_w, q2wbf, (int)SW);

    // 2) three convs as dense WMMA GEMMs, outputs rearranged [b][p][c][l]
    dim3 gConv(HW / 128, COUT / 128, BB);   // (72, 2, 8)
    gemm_bf16_wmma<<<gConv, 256, 0, stream>>>(kwbf,  P1, keyArr, nullptr, 0, 0);
    gemm_bf16_wmma<<<gConv, 256, 0, stream>>>(q1wbf, P1, q1Arr,  nullptr, 0, 0);
    gemm_bf16_wmma<<<gConv, 256, 0, stream>>>(q2wbf, P2, q2Arr,  nullptr, 0, 0);

    // 3) per-(b,p) attention GEMMs + pos bias product
    dim3 gAtt((COUT / 64) * (CIN / 128), 9, BB);   // (8, 9, 8)
    attn_gemm_wmma<<<gAtt, 256, 0, stream>>>(keyArr, q1Arr, q2Arr, pos1, pos2, attnRaw);

    // 4) global mean / unbiased-std normalization, emit bf16 attn
    zero2<<<1, 32, 0, stream>>>(sums);
    reduce_sums<<<1024, 256, 0, stream>>>(attnRaw, NATT, sums);
    normalize_attn<<<2048, 256, 0, stream>>>(attnRaw, sums, km, attnBf, NATT);

    // 5) final per-batch GEMM: attn[256,2304] x im2col(x1)^T -> out f32
    gemm_bf16_wmma<<<gConv, 256, 0, stream>>>(attnBf, P1, nullptr, out, 1, (int)SW);
}
